// Decoder_13572096655605
// MI455X (gfx1250) — matrix operand order repaired and checked by reading, not compiled, not run on hardware
//
#include <hip/hip_runtime.h>
#include <hip/hip_bf16.h>

// Problem constants (match reference)
#define Bb   32
#define Tt   1000
#define Ee   512
#define Mout 80
#define Pp   256
#define Hh   1024
#define G4   (4*Hh)      // 4096 gate width
#define KX   (Pp+Ee)     // 768  lstm0 input width
#define KF   (Hh+Ee)     // 1536 projection input width
#define NROW (Bb*Tt)     // 32000

// LDS staging geometry for recurrent-step weight chunks:
// chunk = 128 K (bf16) x 32 cols x 4 gates; each 256B row padded by 16B so
// 16 rows stride 4 banks apart -> conflict-free ds_load_b128 patterns.
#define BROW   272                 // 256B data + 16B pad
#define BCHUNK (4*32*BROW)         // 34816 B per chunk buffer
#define LDS_STEP (2*BCHUNK + 4*32*32*4)  // 2 chunk buffers + gate LDS = 86016 B

typedef __attribute__((ext_vector_type(16))) __bf16 v16bf;
typedef __attribute__((ext_vector_type(8)))  float  v8f;
typedef __attribute__((ext_vector_type(4)))  unsigned int v4u;
typedef __attribute__((ext_vector_type(8)))  int    v8i;
typedef __attribute__((ext_vector_type(4)))  int    v4i;

union BFTile { v16bf v; uint4 q[2]; };

#if __has_builtin(__builtin_amdgcn_tensor_load_to_lds)
#define USE_TDM 1
#else
#define USE_TDM 0
#endif

#if __has_builtin(__builtin_amdgcn_tanh_f32)
#define TANHF(x) __builtin_amdgcn_tanh_f32(x)
#else
#define TANHF(x) tanhf(x)
#endif

__device__ __forceinline__ v8f wmma_bf16(v16bf a, v16bf b, v8f c) {
  return __builtin_amdgcn_wmma_f32_16x16x32_bf16(false, a, false, b, (short)0, c,
                                                 false, false);
}

__device__ __forceinline__ float sigmoidf_(float x) {
  return 1.0f / (1.0f + __expf(-x));
}

// Deterministic hash-based dropout mask (keep-prob 0.5, scale 2.0).
__device__ __forceinline__ float drop_mask(unsigned idx, unsigned seed) {
  unsigned x = idx * 2654435761u ^ seed;
  x ^= x >> 16; x *= 0x7feb352du;
  x ^= x >> 15; x *= 0x846ca68bu;
  x ^= x >> 16;
  return (x & 1u) ? 2.0f : 0.0f;
}

#if USE_TDM
// TDM: DMA one weight chunk (64 dwords K x 32 cols x 4 gates) into LDS with
// 4-dword padding every 64 dwords (i.e. 16B pad per 256B row).
__device__ __forceinline__ void tdm_stage_b_issue(unsigned lds_off, const __bf16* gsrc) {
  unsigned long long ga = (unsigned long long)gsrc;
  v4u g0;
  g0.x = 1u;                                              // count=1 user D#
  g0.y = lds_off;                                         // LDS byte address
  g0.z = (unsigned)ga;                                    // global_addr[31:0]
  g0.w = (unsigned)((ga >> 32) & 0x1FFFFFFu) | (2u << 30);// addr[56:32] | type=2
  v8i g1;
  g1[0] = (int)((2u << 16) | (1u << 20) | (5u << 22) | (3u << 25));
          // data_size=4B, pad_enable, pad_interval=64dw, pad_amount=4dw
  g1[1] = (int)(512u << 16);          // tensor_dim0 = 512 dw (full row)
  g1[2] = (int)(4096u << 16);         // tensor_dim1 = 4096 rows (lo16)
  g1[3] = (int)(64u << 16);           // tile_dim0 = 64 dw (128 bf16)
  g1[4] = (int)(32u | (4u << 16));    // tile_dim1 = 32 cols, tile_dim2 = 4 gates
  g1[5] = 512;                        // tensor_dim0_stride = 512 dw
  g1[6] = 0;                          // dim1_stride[15:0]<<16 (524288 -> lo16 = 0)
  g1[7] = 8;                          // dim1_stride[47:16] = 524288>>16
  v4i g2 = {4, 0, 0, 0};              // tensor_dim2 = 4
  v4i g3 = {0, 0, 0, 0};
#if defined(__clang_major__) && (__clang_major__ >= 23)
  v8i g4 = {0, 0, 0, 0, 0, 0, 0, 0};
  __builtin_amdgcn_tensor_load_to_lds(g0, g1, g2, g3, g4, 0);
#else
  __builtin_amdgcn_tensor_load_to_lds(g0, g1, g2, g3, 0);
#endif
}
#endif

// Stage one K-chunk of the 4-gate weight slab for this block into LDS.
__device__ __forceinline__ void stage_b(char* dst, const __bf16* W, int cbase,
                                        int koff, int tid) {
#if USE_TDM
  if (tid < 32) {
    tdm_stage_b_issue((unsigned)(unsigned long long)(void*)dst,
                      W + (long)cbase * Hh + koff);
  }
#else
  // Cooperative fallback: 2048 x 16B units, padded LDS rows.
  for (int u = tid; u < 2048; u += 256) {
    const int r = u >> 4, g = r >> 5, ncol = r & 31, iu = u & 15;
    const uint4* s = (const uint4*)(W + ((long)(g * Hh + cbase + ncol)) * Hh + koff) + iu;
    *(uint4*)(dst + r * BROW + iu * 16) = *s;
  }
#endif
}

__device__ __forceinline__ void stage_wait(int tid) {
#if USE_TDM
  if (tid < 32) __builtin_amdgcn_s_wait_tensorcnt(0);
#endif
}

// ---------------------------------------------------------------------------
// 0) One-shot prep: fp32 -> bf16 weights, transpose w_proj, zero LSTM state.
// ---------------------------------------------------------------------------
__global__ void prep_kernel(const float* __restrict__ wih0, const float* __restrict__ whh0,
                            const float* __restrict__ wih1, const float* __restrict__ whh1,
                            const float* __restrict__ wproj,
                            __bf16* __restrict__ wih0b, __bf16* __restrict__ whh0b,
                            __bf16* __restrict__ wih1b, __bf16* __restrict__ whh1b,
                            __bf16* __restrict__ wpT,
                            __bf16* __restrict__ h0A, __bf16* __restrict__ h0B,
                            __bf16* __restrict__ h1A, __bf16* __restrict__ h1B,
                            float* __restrict__ c0, float* __restrict__ c1) {
  long i = (long)blockIdx.x * 256 + threadIdx.x;
  long n;
  n = (long)G4 * KX; if (i < n) { wih0b[i] = (__bf16)wih0[i]; return; } i -= n;
  n = (long)G4 * Hh; if (i < n) { whh0b[i] = (__bf16)whh0[i]; return; } i -= n;
  if (i < n)            { wih1b[i] = (__bf16)wih1[i]; return; } i -= n;
  if (i < n)            { whh1b[i] = (__bf16)whh1[i]; return; } i -= n;
  n = (long)Mout * KF;
  if (i < n) { long col = i / KF, k = i % KF; wpT[i] = (__bf16)wproj[k * Mout + col]; return; }
  i -= n;
  n = (long)Bb * Hh;
  if (i < n) {
    h0A[i] = (__bf16)0.f; h0B[i] = (__bf16)0.f;
    h1A[i] = (__bf16)0.f; h1B[i] = (__bf16)0.f;
    c0[i] = 0.f; c1[i] = 0.f;
  }
}

// ---------------------------------------------------------------------------
// 1) Prenet (80->256->256, relu + deterministic dropout) fused with the
//    teacher-forcing gather: builds X[t*B+b] = [prenet_row | memory_row] bf16.
// ---------------------------------------------------------------------------
__global__ void prenet_build_x(const float* __restrict__ ymels, const float* __restrict__ memory,
                               const float* __restrict__ w1, const float* __restrict__ w2,
                               __bf16* __restrict__ X) {
  const int bt = blockIdx.x;            // = t*Bb + b
  const int t = bt / Bb, b = bt % Bb;
  const int s = (t == 0) ? 0 : (t - 1); // gathered prenet/memory row index
  const int j = threadIdx.x;            // 0..255
  __shared__ float h1[Pp];

  float a1 = 0.f;
  if (s > 0) {                          // y_in[s] = ymels[s-1]; zero row otherwise
    const float* y = ymels + ((long)b * Tt + (s - 1)) * Mout;
    #pragma unroll 4
    for (int k = 0; k < Mout; ++k) a1 = fmaf(y[k], w1[k * Pp + j], a1);
  }
  a1 = fmaxf(a1, 0.f) * drop_mask((unsigned)((b * Tt + s) * Pp + j), 0x9E3779B9u);
  h1[j] = a1;
  __syncthreads();

  float a2 = 0.f;
  #pragma unroll 4
  for (int k = 0; k < Pp; ++k) a2 = fmaf(h1[k], w2[k * Pp + j], a2);
  a2 = fmaxf(a2, 0.f) * drop_mask((unsigned)((b * Tt + s) * Pp + j), 0x85EBCA6Bu);

  __bf16* xr = X + (long)bt * KX;
  xr[j] = (__bf16)a2;
  const float* mr = memory + ((long)b * Tt + s) * Ee;
  xr[Pp + j]       = (__bf16)mr[j];
  xr[Pp + 256 + j] = (__bf16)mr[256 + j];
}

// ---------------------------------------------------------------------------
// 2) Hoisted input GEMM: G0 = X(32000x768) @ w_ih0^T (768x4096) + b0, fp32.
// ---------------------------------------------------------------------------
__global__ void g0_gemm(const __bf16* __restrict__ X, const __bf16* __restrict__ Wb,
                        const float* __restrict__ bih, const float* __restrict__ bhh,
                        float* __restrict__ G0) {
  const int lane = threadIdx.x & 31, w = threadIdx.x >> 5;
  const int hi = lane >> 4, ln = lane & 15;
  const int m0 = blockIdx.x * 16;
  const int n0 = (blockIdx.y * 8 + w) * 16;
  const __bf16* arow = X + (long)(m0 + ln) * KX;
  const __bf16* brow = Wb + (long)(n0 + ln) * KX + hi * 8;
  v8f acc = {};
  #pragma unroll 2
  for (int k0 = 0; k0 < KX; k0 += 32) {
    BFTile A, Bt;
    A.q[0]  = *(const uint4*)(arow + k0 + hi * 8);
    A.q[1]  = *(const uint4*)(arow + k0 + 16 + hi * 8);
    Bt.q[0] = *(const uint4*)(brow + k0);
    Bt.q[1] = *(const uint4*)(brow + k0 + 16);
    acc = wmma_bf16(A.v, Bt.v, acc);
  }
  #pragma unroll
  for (int v = 0; v < 8; ++v) {
    const int m = m0 + v + 8 * hi;
    const int n = n0 + ln;
    G0[(long)m * G4 + n] = acc[v] + bih[n] + bhh[n];
  }
}

// ---------------------------------------------------------------------------
// 3a) LSTM layer-0 step: gates = G0[t] + h0 @ w_hh0^T, then nonlinearity.
//     32 blocks of 32 hidden cols; wave = (gate, m-half), 2 N-tiles per wave.
//     Weights TDM-staged into LDS (double buffered), B fed via ds_load_b128.
// ---------------------------------------------------------------------------
__global__ void lstm_l0_step(int t, const float* __restrict__ G0,
                             const __bf16* __restrict__ h_in, __bf16* __restrict__ h_out,
                             float* __restrict__ c, const __bf16* __restrict__ Whh) {
  extern __shared__ char smem[];
  char* bb0 = smem;
  char* bb1 = smem + BCHUNK;
  float (*gsum)[32][32] = (float (*)[32][32])(smem + 2 * BCHUNK);

  const int tid = threadIdx.x;
  const int lane = tid & 31, w = tid >> 5;
  const int hi = lane >> 4, ln = lane & 15;
  const int gate = w >> 1, mh = w & 1;
  const int cbase = blockIdx.x * 32;

  const __bf16* arow = h_in + (long)(mh * 16 + ln) * Hh + hi * 16;
  const char* blds0 = (const char*)0;

  stage_b(bb0, Whh, cbase, 0, tid);               // chunk 0 in flight
  v8f acc0 = {}, acc1 = {};
  for (int cc = 0; cc < 8; ++cc) {
    stage_wait(tid);
    __syncthreads();                               // chunk cc visible to all
    if (cc + 1 < 8) stage_b((cc & 1) ? bb0 : bb1, Whh, cbase, (cc + 1) * 128, tid);
    const char* bb = (cc & 1) ? bb1 : bb0;
    const char* bl0 = bb + (gate * 32 + ln) * BROW + hi * 32;
    const char* bl1 = bb + (gate * 32 + 16 + ln) * BROW + hi * 32;
    #pragma unroll
    for (int k4 = 0; k4 < 4; ++k4) {
      const int k0 = cc * 128 + k4 * 32;
      BFTile A, B0, B1;
      A.q[0]  = *(const uint4*)(arow + k0);
      A.q[1]  = *(const uint4*)(arow + k0 + 8);
      B0.q[0] = *(const uint4*)(bl0 + k4 * 64);
      B0.q[1] = *(const uint4*)(bl0 + k4 * 64 + 16);
      B1.q[0] = *(const uint4*)(bl1 + k4 * 64);
      B1.q[1] = *(const uint4*)(bl1 + k4 * 64 + 16);
      acc0 = wmma_bf16(A.v, B0.v, acc0);
      acc1 = wmma_bf16(A.v, B1.v, acc1);
    }
    __syncthreads();                               // done reading chunk cc
  }
  (void)blds0;

  #pragma unroll
  for (int v = 0; v < 8; ++v) {
    const int ml = mh * 16 + v + 8 * hi;
    gsum[gate][ml][ln]      = acc0[v];
    gsum[gate][ml][16 + ln] = acc1[v];
  }
  __syncthreads();

  #pragma unroll
  for (int e = 0; e < 4; ++e) {
    const int idx = tid * 4 + e;                   // 0..1023 over 32x32 elements
    const int m = idx >> 5, nl = idx & 31;
    const int col = cbase + nl;
    const float* grow = G0 + (long)(t * Bb + m) * G4;
    const float gi = gsum[0][m][nl] + grow[col];
    const float gf = gsum[1][m][nl] + grow[Hh + col];
    const float gg = gsum[2][m][nl] + grow[2 * Hh + col];
    const float go = gsum[3][m][nl] + grow[3 * Hh + col];
    const float cp = c[(long)m * Hh + col];
    const float cn = sigmoidf_(gf) * cp + sigmoidf_(gi) * TANHF(gg);
    c[(long)m * Hh + col] = cn;
    h_out[(long)m * Hh + col] = (__bf16)(sigmoidf_(go) * TANHF(cn));
  }
}

// ---------------------------------------------------------------------------
// 3b) LSTM layer-1 step: gates = h0 @ w_ih1^T + h1 @ w_hh1^T + b1.
//     16 TDM chunks: chunks 0..7 = w_ih1 with A=h0, 8..15 = w_hh1 with A=h1.
// ---------------------------------------------------------------------------
__global__ void lstm_l1_step(int t, const __bf16* __restrict__ h0cur,
                             const __bf16* __restrict__ h_in, __bf16* __restrict__ h_out,
                             float* __restrict__ c,
                             const __bf16* __restrict__ Wih, const __bf16* __restrict__ Whh,
                             const float* __restrict__ bih, const float* __restrict__ bhh,
                             float* __restrict__ hs) {
  extern __shared__ char smem[];
  char* bb0 = smem;
  char* bb1 = smem + BCHUNK;
  float (*gsum)[32][32] = (float (*)[32][32])(smem + 2 * BCHUNK);

  const int tid = threadIdx.x;
  const int lane = tid & 31, w = tid >> 5;
  const int hi = lane >> 4, ln = lane & 15;
  const int gate = w >> 1, mh = w & 1;
  const int cbase = blockIdx.x * 32;

  const __bf16* a0 = h0cur + (long)(mh * 16 + ln) * Hh + hi * 16;
  const __bf16* a1 = h_in  + (long)(mh * 16 + ln) * Hh + hi * 16;

  stage_b(bb0, Wih, cbase, 0, tid);
  v8f acc0 = {}, acc1 = {};
  for (int cc = 0; cc < 16; ++cc) {
    stage_wait(tid);
    __syncthreads();
    if (cc + 1 < 16) {
      const __bf16* Wn = (cc + 1 < 8) ? Wih : Whh;
      stage_b((cc & 1) ? bb0 : bb1, Wn, cbase, ((cc + 1) & 7) * 128, tid);
    }
    const char* bb = (cc & 1) ? bb1 : bb0;
    const char* bl0 = bb + (gate * 32 + ln) * BROW + hi * 32;
    const char* bl1 = bb + (gate * 32 + 16 + ln) * BROW + hi * 32;
    const __bf16* ar = (cc < 8) ? a0 : a1;
    #pragma unroll
    for (int k4 = 0; k4 < 4; ++k4) {
      const int k0 = (cc & 7) * 128 + k4 * 32;
      BFTile A, B0, B1;
      A.q[0]  = *(const uint4*)(ar + k0);
      A.q[1]  = *(const uint4*)(ar + k0 + 8);
      B0.q[0] = *(const uint4*)(bl0 + k4 * 64);
      B0.q[1] = *(const uint4*)(bl0 + k4 * 64 + 16);
      B1.q[0] = *(const uint4*)(bl1 + k4 * 64);
      B1.q[1] = *(const uint4*)(bl1 + k4 * 64 + 16);
      acc0 = wmma_bf16(A.v, B0.v, acc0);
      acc1 = wmma_bf16(A.v, B1.v, acc1);
    }
    __syncthreads();
  }

  #pragma unroll
  for (int v = 0; v < 8; ++v) {
    const int ml = mh * 16 + v + 8 * hi;
    gsum[gate][ml][ln]      = acc0[v];
    gsum[gate][ml][16 + ln] = acc1[v];
  }
  __syncthreads();

  #pragma unroll
  for (int e = 0; e < 4; ++e) {
    const int idx = tid * 4 + e;
    const int m = idx >> 5, nl = idx & 31;
    const int col = cbase + nl;
    const float gi = gsum[0][m][nl] + bih[col] + bhh[col];
    const float gf = gsum[1][m][nl] + bih[Hh + col] + bhh[Hh + col];
    const float gg = gsum[2][m][nl] + bih[2 * Hh + col] + bhh[2 * Hh + col];
    const float go = gsum[3][m][nl] + bih[3 * Hh + col] + bhh[3 * Hh + col];
    const float cp = c[(long)m * Hh + col];
    const float cn = sigmoidf_(gf) * cp + sigmoidf_(gi) * TANHF(gg);
    c[(long)m * Hh + col] = cn;
    const float hv = sigmoidf_(go) * TANHF(cn);
    h_out[(long)m * Hh + col] = (__bf16)hv;
    hs[((long)m * Tt + t) * Hh + col] = hv;        // (B,T,H) for the projection
  }
}

// ---------------------------------------------------------------------------
// 4) Projection: out = [hs | memory](32000x1536) @ w_projT + b_proj.
// ---------------------------------------------------------------------------
__global__ void proj_gemm(const float* __restrict__ hs, const float* __restrict__ memory,
                          const __bf16* __restrict__ WpT, const float* __restrict__ bproj,
                          float* __restrict__ out) {
  const int lane = threadIdx.x & 31, w = threadIdx.x >> 5;
  const int hi = lane >> 4, ln = lane & 15;
  const int m0 = (blockIdx.x * 8 + w) * 16;
  const long r = m0 + ln;                 // A row = b*T + t for both sources
  v8f acc[5] = {};
  for (int k0 = 0; k0 < KF; k0 += 32) {
    const float* ar = (k0 < Hh) ? (hs + r * Hh + k0) : (memory + r * Ee + (k0 - Hh));
    BFTile A;
    #pragma unroll
    for (int c8 = 0; c8 < 2; ++c8) {
      const float* p = ar + hi * 16 + c8 * 8;
      const float4 f0 = *(const float4*)(p);
      const float4 f1 = *(const float4*)(p + 4);
      A.v[c8 * 8 + 0] = (__bf16)f0.x; A.v[c8 * 8 + 1] = (__bf16)f0.y;
      A.v[c8 * 8 + 2] = (__bf16)f0.z; A.v[c8 * 8 + 3] = (__bf16)f0.w;
      A.v[c8 * 8 + 4] = (__bf16)f1.x; A.v[c8 * 8 + 5] = (__bf16)f1.y;
      A.v[c8 * 8 + 6] = (__bf16)f1.z; A.v[c8 * 8 + 7] = (__bf16)f1.w;
    }
    #pragma unroll
    for (int nt = 0; nt < 5; ++nt) {
      const __bf16* brow = WpT + (long)(nt * 16 + ln) * KF + k0 + hi * 16;
      BFTile Bt;
      Bt.q[0] = *(const uint4*)(brow);
      Bt.q[1] = *(const uint4*)(brow + 8);
      acc[nt] = wmma_bf16(A.v, Bt.v, acc[nt]);
    }
  }
  #pragma unroll
  for (int nt = 0; nt < 5; ++nt)
    #pragma unroll
    for (int v = 0; v < 8; ++v) {
      const int m = m0 + v + 8 * hi;
      const int n = nt * 16 + ln;
      out[(long)m * Mout + n] = acc[nt][v] + bproj[n];
    }
}

// ---------------------------------------------------------------------------
extern "C" void kernel_launch(void* const* d_in, const int* in_sizes, int n_in,
                              void* d_out, int out_size, void* d_ws, size_t ws_size,
                              hipStream_t stream) {
  const float* memory = (const float*)d_in[0];
  const float* ymels  = (const float*)d_in[1];
  const float* w_pre1 = (const float*)d_in[2];
  const float* w_pre2 = (const float*)d_in[3];
  const float* w_ih0  = (const float*)d_in[4];
  const float* w_hh0  = (const float*)d_in[5];
  const float* b_ih0  = (const float*)d_in[6];
  const float* b_hh0  = (const float*)d_in[7];
  const float* w_ih1  = (const float*)d_in[8];
  const float* w_hh1  = (const float*)d_in[9];
  const float* b_ih1  = (const float*)d_in[10];
  const float* b_hh1  = (const float*)d_in[11];
  const float* w_proj = (const float*)d_in[12];
  const float* b_proj = (const float*)d_in[13];
  float* out = (float*)d_out;

  char* ws = (char*)d_ws;
  size_t off = 0;
  auto take = [&](size_t bytes) -> void* {
    void* p = (void*)(ws + off);
    off += (bytes + 255) & ~(size_t)255;
    return p;
  };
  __bf16* X     = (__bf16*)take((size_t)NROW * KX * 2);
  float*  G0    = (float*) take((size_t)NROW * G4 * 4);
  float*  hs    = (float*) take((size_t)NROW * Hh * 4);
  __bf16* wih0b = (__bf16*)take((size_t)G4 * KX * 2);
  __bf16* whh0b = (__bf16*)take((size_t)G4 * Hh * 2);
  __bf16* wih1b = (__bf16*)take((size_t)G4 * Hh * 2);
  __bf16* whh1b = (__bf16*)take((size_t)G4 * Hh * 2);
  __bf16* wpT   = (__bf16*)take((size_t)Mout * KF * 2);
  __bf16* h0A   = (__bf16*)take((size_t)Bb * Hh * 2);
  __bf16* h0B   = (__bf16*)take((size_t)Bb * Hh * 2);
  __bf16* h1A   = (__bf16*)take((size_t)Bb * Hh * 2);
  __bf16* h1B   = (__bf16*)take((size_t)Bb * Hh * 2);
  float*  c0    = (float*) take((size_t)Bb * Hh * 4);
  float*  c1    = (float*) take((size_t)Bb * Hh * 4);
  if (off > ws_size) return;   // workspace too small; refuse to scribble OOB

  const long prep_elems = (long)G4 * KX + 3L * G4 * Hh + (long)Mout * KF + (long)Bb * Hh;
  const int prep_blocks = (int)((prep_elems + 255) / 256);
  prep_kernel<<<prep_blocks, 256, 0, stream>>>(w_ih0, w_hh0, w_ih1, w_hh1, w_proj,
                                               wih0b, whh0b, wih1b, whh1b, wpT,
                                               h0A, h0B, h1A, h1B, c0, c1);
  prenet_build_x<<<NROW, 256, 0, stream>>>(ymels, memory, w_pre1, w_pre2, X);
  g0_gemm<<<dim3(NROW / 16, 32), 256, 0, stream>>>(X, wih0b, b_ih0, b_hh0, G0);

  for (int t = 0; t < Tt; ++t) {
    const __bf16* h0i = (t & 1) ? h0B : h0A;  __bf16* h0o = (t & 1) ? h0A : h0B;
    const __bf16* h1i = (t & 1) ? h1B : h1A;  __bf16* h1o = (t & 1) ? h1A : h1B;
    lstm_l0_step<<<Hh / 32, 256, LDS_STEP, stream>>>(t, G0, h0i, h0o, c0, whh0b);
    lstm_l1_step<<<Hh / 32, 256, LDS_STEP, stream>>>(t, h0o, h1i, h1o, c1,
                                                     wih1b, whh1b, b_ih1, b_hh1, hs);
  }

  proj_gemm<<<NROW / 16 / 8, 256, 0, stream>>>(hs, memory, wpT, b_proj, out);
}
